// MultiHeadAttention_77910706749686
// MI455X (gfx1250) — matrix operand order repaired and checked by reading, not compiled, not run on hardware
//
#include <hip/hip_runtime.h>
#include <hip/hip_bf16.h>

// ---------------------------------------------------------------------------
// Multi-head attention for MI455X (gfx1250), f16 WMMA with f32 accumulation.
// Pipeline: prep_weights -> proj (Q/K/V heads, V stored transposed) ->
//           flash attention (async-LDS K staging, online softmax) -> out proj.
// ---------------------------------------------------------------------------

typedef __attribute__((ext_vector_type(16))) _Float16 v16h;
typedef __attribute__((ext_vector_type(8)))  _Float16 v8h;
typedef __attribute__((ext_vector_type(8)))  float    v8f;
typedef int vi4 __attribute__((vector_size(16)));    // int4 for async-LDS ops

#define B_   4
#define S_   2048
#define D_   1024
#define H_   16
#define DK_  64
#define DV_  64
#define E_   (H_ * DV_)     // 1024

// workspace offsets (in _Float16 elements)
#define WQT_OFF  ((size_t)0)                       // [H][DK][D]   1 M
#define WKT_OFF  ((size_t)1048576)                 // [H][DK][D]   1 M
#define WVT_OFF  ((size_t)2097152)                 // [H][DV][D]   1 M
#define WOT_OFF  ((size_t)3145728)                 // [D][E]       1 M
#define QH_OFF   ((size_t)4194304)                 // [B][H][S][DK] 8 M
#define KH_OFF   ((size_t)12582912)                // [B][H][S][DK] 8 M
#define VHT_OFF  ((size_t)20971520)                // [B][H][DV][S] 8 M (transposed)
#define MIX_OFF  ((size_t)29360128)                // [B][S][E]     8 M

// CDNA5 async global->LDS path (ASYNCcnt), if the toolchain exposes it.
#if defined(__has_builtin)
#if __has_builtin(__builtin_amdgcn_global_load_async_to_lds_b128) && \
    __has_builtin(__builtin_amdgcn_s_wait_asynccnt)
#define HAVE_ASYNC_LDS 1
#endif
#endif
#ifndef HAVE_ASYNC_LDS
#define HAVE_ASYNC_LDS 0
#endif

// Compile-time probe: report which K-staging path was built (shows in stderr).
#if HAVE_ASYNC_LDS
#warning "CDNA5 probe: ASYNC-LDS path ENABLED (global_load_async_to_lds_b128 + s_wait_asynccnt)"
#else
#warning "CDNA5 probe: ASYNC-LDS path DISABLED (falling back to direct global loads)"
#endif

#if HAVE_ASYNC_LDS
typedef __attribute__((address_space(1))) vi4 gvi4;   // global int4
typedef __attribute__((address_space(3))) vi4 svi4;   // LDS int4
#endif

__device__ inline v8f wmma16(v16h a, v16h b, v8f c) {
    return __builtin_amdgcn_wmma_f32_16x16x32_f16(
        false, a, false, b, (short)0, c, false, false);
}

// Build 16x32 f16 A-fragment from two 8-half contiguous runs (f16 source).
__device__ inline v16h a_frag_f16(const _Float16* p0, const _Float16* p1) {
    v8h lo = *(const v8h*)p0;
    v8h hi = *(const v8h*)p1;
    v16h a;
#pragma unroll
    for (int i = 0; i < 8; ++i) { a[i] = lo[i]; a[8 + i] = hi[i]; }
    return a;
}

// Build 16x32 f16 A-fragment from f32 source (convert on the fly).
__device__ inline v16h a_frag_f32(const float* p0, const float* p1) {
    v16h a;
#pragma unroll
    for (int i = 0; i < 8; ++i) {
        a[i]     = (_Float16)p0[i];
        a[8 + i] = (_Float16)p1[i];
    }
    return a;
}

// ---------------------------------------------------------------------------
// Kernel 1: convert + transpose weights to f16 (blockIdx.y selects matrix).
// All dims are powers of two -> pure shift/mask indexing.
// ---------------------------------------------------------------------------
__global__ __launch_bounds__(256) void prep_weights(
    const float* __restrict__ Wq, const float* __restrict__ Wk,
    const float* __restrict__ Wv, const float* __restrict__ Wo,
    _Float16* __restrict__ ws) {
    unsigned idx = blockIdx.x * 256u + threadIdx.x;   // 0 .. 2^20-1
    unsigned mat = blockIdx.y;
    if (mat < 3u) {                                   // W[h][d][c] -> [h][c][d]
        unsigned h = idx >> 16, d = (idx >> 6) & 1023u, c = idx & 63u;
        const float* W = (mat == 0u) ? Wq : (mat == 1u) ? Wk : Wv;
        size_t off = (mat == 0u) ? WQT_OFF : (mat == 1u) ? WKT_OFF : WVT_OFF;
        ws[off + (((size_t)(h * 64u + c)) << 10) + d] = (_Float16)W[idx];
    } else {                                          // Wo[e][d] -> [d][e]
        unsigned e = idx >> 10, d = idx & 1023u;
        ws[WOT_OFF + (((size_t)d) << 10) + e] = (_Float16)Wo[idx];
    }
}

// ---------------------------------------------------------------------------
// Kernel 2: per-head projections. One wave -> 16 tokens x full head (64 cols).
// mat 0: Qh[b,h,s,c]; mat 1: Kh[b,h,s,c]; mat 2: Vh transposed [b,h,c,s].
// ---------------------------------------------------------------------------
__global__ __launch_bounds__(256) void proj_kernel(
    const float* __restrict__ q, const float* __restrict__ k,
    const float* __restrict__ v, _Float16* __restrict__ ws) {
    unsigned wid  = blockIdx.x * 8u + (threadIdx.x >> 5);
    unsigned lane = threadIdx.x & 31u;
    unsigned mat   = wid >> 13;          // / (512 tiles * 16 heads)
    unsigned rem   = wid & 8191u;
    unsigned ttile = rem >> 4;
    unsigned h     = rem & 15u;
    unsigned t0    = ttile * 16u;

    const float* src = (mat == 0u) ? q : ((mat == 1u) ? k : v);
    const _Float16* wT = ws +
        ((mat == 0u) ? WQT_OFF : (mat == 1u) ? WKT_OFF : WVT_OFF) +
        (size_t)h * DK_ * D_;

    unsigned row   = lane & 15u;
    unsigned abase = (lane >> 4) * 16u;
    unsigned kb    = (lane >> 4) * 16u;

    v8f acc[4] = {};
    for (unsigned kc = 0; kc < D_ / 32; ++kc) {
        unsigned d0 = kc * 32u;
        const float* ap = src + (size_t)(t0 + row) * D_ + d0 + abase;
        v16h a = a_frag_f32(ap, ap + 8);
#pragma unroll
        for (int ct = 0; ct < 4; ++ct) {
            const _Float16* bp = wT + (size_t)(ct * 16u + (lane & 15u)) * D_ + d0 + kb;
            acc[ct] = wmma16(a, *(const v16h*)bp, acc[ct]);
        }
    }

    // store epilogue: batch index is uniform per tile (tiles never cross S).
    unsigned bb   = t0 >> 11;            // t0 / S_
    unsigned sI   = t0 & 2047u;          // t0 % S_
    unsigned mtop = (lane >> 4) * 8u;

    if (mat < 2u) {
        _Float16* dst = ws + ((mat == 0u) ? QH_OFF : KH_OFF) +
            (((size_t)(bb * H_ + h)) * S_ + sI + mtop) * DK_ + (lane & 15u);
#pragma unroll
        for (int ct = 0; ct < 4; ++ct)
#pragma unroll
            for (int rr = 0; rr < 8; ++rr)
                dst[(size_t)rr * DK_ + ct * 16] = (_Float16)acc[ct][rr];
    } else {
        // transposed V: consecutive rows are contiguous -> packed 16B stores
        _Float16* dst = ws + VHT_OFF +
            (((size_t)(bb * H_ + h)) * DV_ + (lane & 15u)) * S_ + sI + mtop;
#pragma unroll
        for (int ct = 0; ct < 4; ++ct) {
            v8h pk;
#pragma unroll
            for (int rr = 0; rr < 8; ++rr) pk[rr] = (_Float16)acc[ct][rr];
            *(v8h*)(dst + (size_t)(ct * 16) * S_) = pk;
        }
    }
}

// ---------------------------------------------------------------------------
// Kernel 3: flash attention. One wave per 16-query tile per (b,h).
// 128-thread blocks (4 waves); K tiles staged to LDS via async loads.
// ---------------------------------------------------------------------------
__global__ __launch_bounds__(128) void attn_kernel(
    const unsigned char* __restrict__ mask, _Float16* __restrict__ ws) {
    __shared__ __align__(16) _Float16 Pb[4][16 * 32];        // P staging, 4 KB
#if HAVE_ASYNC_LDS
    __shared__ __align__(16) _Float16 Kst[4][2][32 * DK_];   // K dbl buf, 32 KB
#endif

    unsigned widx = threadIdx.x >> 5;
    unsigned lane = threadIdx.x & 31u;
    unsigned wg   = blockIdx.x * 4u + widx;
    unsigned qt = wg & 127u;             // S_/16 = 128
    unsigned bh = wg >> 7;
    unsigned h  = bh & 15u;
    unsigned b  = bh >> 4;
    unsigned q0 = qt * 16u;

    const _Float16* Qp = ws + QH_OFF + (((size_t)(b * H_ + h)) * S_ + q0) * DK_;
    const _Float16* Kp = ws + KH_OFF + ((size_t)(b * H_ + h)) * S_ * DK_;
    const _Float16* Vp = ws + VHT_OFF + ((size_t)(b * H_ + h)) * DV_ * S_;
    const unsigned char* mrow = mask + (size_t)b * S_;

    unsigned row   = lane & 15u;
    unsigned abase = (lane >> 4) * 16u;
    unsigned kb    = (lane >> 4) * 16u;

    v16h aq0 = a_frag_f16(Qp + row * DK_ + abase,      Qp + row * DK_ + abase + 8);
    v16h aq1 = a_frag_f16(Qp + row * DK_ + abase + 32, Qp + row * DK_ + abase + 40);

    v8f o0 = {}, o1 = {}, o2 = {}, o3 = {};
    float mi[8], li[8];
#pragma unroll
    for (int rr = 0; rr < 8; ++rr) { mi[rr] = -1e30f; li[rr] = 0.0f; }

    _Float16* P = &Pb[widx][0];

#if HAVE_ASYNC_LDS
    // stage one 32-key x 64-dk f16 block (contiguous 4 KB) into LDS buffer
    auto stage_k = [&](unsigned s0, _Float16* dst) {
        const char* g = (const char*)(Kp + (size_t)s0 * DK_) + lane * 16u;
        char*       l = (char*)dst + lane * 16u;
#pragma unroll
        for (int i = 0; i < 8; ++i) {
            __builtin_amdgcn_global_load_async_to_lds_b128(
                (gvi4*)(g + i * 512), (svi4*)(l + i * 512), 0, 0);
        }
    };
    unsigned buf = 0;
    stage_k(0u, &Kst[widx][0][0]);
#endif

    for (unsigned s0 = 0; s0 < S_; s0 += 32) {
        if (s0 + 32 < S_) {  // prefetch next V block (global_prefetch_b8)
            __builtin_prefetch(Vp + (size_t)lane * S_ + (s0 + 32), 0, 1);
        }
#if HAVE_ASYNC_LDS
        if (s0 + 32 < S_) {
            stage_k(s0 + 32, &Kst[widx][buf ^ 1u][0]);
            __builtin_amdgcn_s_wait_asynccnt(8);   // current block landed
        } else {
            __builtin_amdgcn_s_wait_asynccnt(0);
        }
        const _Float16* KW = &Kst[widx][buf][0];
        v16h bk00 = *(const v16h*)(KW + (size_t)(lane & 15u) * DK_ + kb);
        v16h bk01 = *(const v16h*)(KW + (size_t)(lane & 15u) * DK_ + 32 + kb);
        v16h bk10 = *(const v16h*)(KW + (size_t)(16u + (lane & 15u)) * DK_ + kb);
        v16h bk11 = *(const v16h*)(KW + (size_t)(16u + (lane & 15u)) * DK_ + 32 + kb);
        buf ^= 1u;
#else
        unsigned key0g = s0 + (lane & 15u);
        v16h bk00 = *(const v16h*)(Kp + (size_t)key0g * DK_ + kb);
        v16h bk01 = *(const v16h*)(Kp + (size_t)key0g * DK_ + 32 + kb);
        v16h bk10 = *(const v16h*)(Kp + (size_t)(key0g + 16u) * DK_ + kb);
        v16h bk11 = *(const v16h*)(Kp + (size_t)(key0g + 16u) * DK_ + 32 + kb);
#endif

        v8f c0 = {}, c1 = {};
        c0 = wmma16(aq0, bk00, c0); c0 = wmma16(aq1, bk01, c0);
        c1 = wmma16(aq0, bk10, c1); c1 = wmma16(aq1, bk11, c1);

        bool ok0 = mrow[s0 + (lane & 15u)] != 0;
        bool ok1 = mrow[s0 + 16u + (lane & 15u)] != 0;

#pragma unroll
        for (int rr = 0; rr < 8; ++rr) {
            float s0v = ok0 ? c0[rr] * 0.125f : -1e30f;   // 1/sqrt(64)
            float s1v = ok1 ? c1[rr] * 0.125f : -1e30f;
            float mx = fmaxf(s0v, s1v);
            mx = fmaxf(mx, __shfl_xor(mx, 1));
            mx = fmaxf(mx, __shfl_xor(mx, 2));
            mx = fmaxf(mx, __shfl_xor(mx, 4));
            mx = fmaxf(mx, __shfl_xor(mx, 8));
            float nm    = fmaxf(mi[rr], mx);
            float alpha = __expf(mi[rr] - nm);
            float p0 = __expf(s0v - nm);
            float p1 = __expf(s1v - nm);
            float rs = p0 + p1;
            rs += __shfl_xor(rs, 1);
            rs += __shfl_xor(rs, 2);
            rs += __shfl_xor(rs, 4);
            rs += __shfl_xor(rs, 8);
            li[rr] = li[rr] * alpha + rs;
            mi[rr] = nm;
            unsigned m = (unsigned)rr + ((lane >> 4) << 3);
            P[m * 32 + (lane & 15u)]      = (_Float16)p0;
            P[m * 32 + 16 + (lane & 15u)] = (_Float16)p1;
            o0[rr] *= alpha; o1[rr] *= alpha; o2[rr] *= alpha; o3[rr] *= alpha;
        }
        // reload P in A-fragment layout (same-wave LDS ordering)
        v16h pa = a_frag_f16(P + row * 32 + abase, P + row * 32 + abase + 8);

        o0 = wmma16(pa, *(const v16h*)(Vp + (size_t)((lane & 15u))      * S_ + s0 + kb), o0);
        o1 = wmma16(pa, *(const v16h*)(Vp + (size_t)(16u + (lane & 15u)) * S_ + s0 + kb), o1);
        o2 = wmma16(pa, *(const v16h*)(Vp + (size_t)(32u + (lane & 15u)) * S_ + s0 + kb), o2);
        o3 = wmma16(pa, *(const v16h*)(Vp + (size_t)(48u + (lane & 15u)) * S_ + s0 + kb), o3);
    }

    unsigned mtop = (lane >> 4) << 3;
    _Float16* mixp = ws + MIX_OFF + ((size_t)b * S_ + q0 + mtop) * E_ +
                     h * DV_ + (lane & 15u);
#pragma unroll
    for (int rr = 0; rr < 8; ++rr) {
        float inv = 1.0f / li[rr];
        mixp[(size_t)rr * E_ +  0] = (_Float16)(o0[rr] * inv);
        mixp[(size_t)rr * E_ + 16] = (_Float16)(o1[rr] * inv);
        mixp[(size_t)rr * E_ + 32] = (_Float16)(o2[rr] * inv);
        mixp[(size_t)rr * E_ + 48] = (_Float16)(o3[rr] * inv);
    }
}

// ---------------------------------------------------------------------------
// Kernel 4: output projection mix[B*S,E] x Wo[E,D] -> out f32 [B*S,D].
// ---------------------------------------------------------------------------
__global__ __launch_bounds__(256) void outproj_kernel(
    const _Float16* __restrict__ ws, float* __restrict__ out) {
    unsigned wid  = blockIdx.x * 8u + (threadIdx.x >> 5);
    unsigned lane = threadIdx.x & 31u;
    unsigned ttile = wid >> 4;           // 512 token tiles
    unsigned dg    = wid & 15u;          // 16 groups of 64 output cols
    unsigned t0 = ttile * 16u;
    unsigned d0 = dg * 64u;

    const _Float16* mixp = ws + MIX_OFF;
    const _Float16* WoTp = ws + WOT_OFF;

    unsigned row   = lane & 15u;
    unsigned abase = (lane >> 4) * 16u;
    unsigned kb    = (lane >> 4) * 16u;

    v8f acc[4] = {};
    for (unsigned ec = 0; ec < E_ / 32; ++ec) {
        unsigned e0 = ec * 32u;
        const _Float16* ap = mixp + (size_t)(t0 + row) * E_ + e0 + abase;
        v16h a = a_frag_f16(ap, ap + 8);
#pragma unroll
        for (int ct = 0; ct < 4; ++ct) {
            unsigned col = d0 + ct * 16u + (lane & 15u);
            const _Float16* bp = WoTp + (size_t)col * E_ + e0 + kb;
            acc[ct] = wmma16(a, *(const v16h*)bp, acc[ct]);
        }
    }

    unsigned mtop = (lane >> 4) << 3;
    float* dst = out + (size_t)(t0 + mtop) * D_ + d0 + (lane & 15u);
#pragma unroll
    for (int ct = 0; ct < 4; ++ct)
#pragma unroll
        for (int rr = 0; rr < 8; ++rr)
            dst[(size_t)rr * D_ + ct * 16] = acc[ct][rr];
}

// ---------------------------------------------------------------------------
extern "C" void kernel_launch(void* const* d_in, const int* in_sizes, int n_in,
                              void* d_out, int out_size, void* d_ws, size_t ws_size,
                              hipStream_t stream) {
    const float* q  = (const float*)d_in[0];
    const float* k  = (const float*)d_in[1];
    const float* v  = (const float*)d_in[2];
    const unsigned char* mask = (const unsigned char*)d_in[3];
    const float* Wq = (const float*)d_in[4];
    const float* Wk = (const float*)d_in[5];
    const float* Wv = (const float*)d_in[6];
    const float* Wo = (const float*)d_in[7];
    _Float16* ws = (_Float16*)d_ws;
    float* out = (float*)d_out;

    // 1) weight convert/transpose: 4 matrices x 1M elements
    prep_weights<<<dim3(4096, 4), 256, 0, stream>>>(Wq, Wk, Wv, Wo, ws);
    // 2) projections: 3 mats * 512 token tiles * 16 heads = 24576 waves
    proj_kernel<<<3072, 256, 0, stream>>>(q, k, v, ws);
    // 3) flash attention: 4*16*128 = 8192 waves, 4 waves/block
    attn_kernel<<<2048, 128, 0, stream>>>(mask, ws);
    // 4) output projection: 512*16 = 8192 waves
    outproj_kernel<<<1024, 256, 0, stream>>>(ws, out);
}
